// Decoder_60636348285399
// MI455X (gfx1250) — matrix operand order repaired and checked by reading, not compiled, not run on hardware
//
#include <hip/hip_runtime.h>
#include <math.h>

// ---------------------------------------------------------------------------
// Types for CDNA5 WMMA (wave32). bf16 16x16x32 -> f32 accum.
// ---------------------------------------------------------------------------
typedef __attribute__((ext_vector_type(16))) __bf16   v16bf;
typedef __attribute__((ext_vector_type(8)))  float    v8f;
typedef __attribute__((ext_vector_type(4)))  unsigned u32x4;

__device__ __forceinline__ unsigned short f2bf(float f) {
    unsigned x = __builtin_bit_cast(unsigned, f);
    unsigned r = x + 0x7FFFu + ((x >> 16) & 1u);   // round-to-nearest-even
    return (unsigned short)(r >> 16);
}

__device__ __forceinline__ float sigmoidf_(float x) {
    return 1.0f / (1.0f + __expf(-x));
}

// A-matrix fragment (16-bit, 16x32): lane m = l&15, half = l>>4 holds
// K = [8h, 8h+8) and [16+8h, 16+8h+8)  (two 16B chunks)
__device__ __forceinline__ v16bf load_a_frag(const unsigned short* row, int kk, int half) {
    union { u32x4 u[2]; v16bf v; } p;
    p.u[0] = *reinterpret_cast<const u32x4*>(row + kk + 8 * half);
    p.u[1] = *reinterpret_cast<const u32x4*>(row + kk + 16 + 8 * half);
    return p.v;
}
// B-matrix fragment (16-bit, 32x16): lane n = l&15, half = l>>4 holds
// K = [8h, 8h+8) and [16+8h, 16+8h+8)  (two 16B chunks)
__device__ __forceinline__ v16bf load_b_frag(const unsigned short* row, int kk, int half) {
    union { u32x4 u[2]; v16bf v; } p;
    p.u[0] = *reinterpret_cast<const u32x4*>(row + kk + 8 * half);
    p.u[1] = *reinterpret_cast<const u32x4*>(row + kk + 16 + 8 * half);
    return p.v;
}

// ---------------------------------------------------------------------------
// Workspace layout (bytes, all 256B aligned)
// ---------------------------------------------------------------------------
static constexpr size_t SZ_WA   = (size_t)4096 * 1792 * 2;   // att [Wih|Whh] bf16
static constexpr size_t SZ_WD   = (size_t)4096 * 2560 * 2;   // dec [Wih|Whh] bf16
static constexpr size_t SZ_WP   = (size_t)96   * 1536 * 2;   // proj(80)+gate(1)+pad bf16
static constexpr size_t SZ_XS   = (size_t)1000 * 64 * 256 * 2; // prenet out bf16
static constexpr size_t SZ_PM   = (size_t)64 * 512 * 128 * 4;  // processed_memory f32
static constexpr size_t SZ_UATT = (size_t)64 * 1792 * 2;     // [x|ctx|h_att] bf16
static constexpr size_t SZ_UDEC = (size_t)64 * 2560 * 2;     // [h_att|ctx|h_dec] bf16
static constexpr size_t SZ_HC   = (size_t)64 * 1536 * 2;     // [h_dec|ctx] bf16
static constexpr size_t SZ_G    = (size_t)64 * 4096 * 4;     // gates f32
static constexpr size_t SZ_ST   = (size_t)64 * 1024 * 4;     // one f32 state

static constexpr size_t OFF_WA   = 0;
static constexpr size_t OFF_WD   = OFF_WA   + SZ_WA;
static constexpr size_t OFF_WP   = OFF_WD   + SZ_WD;
static constexpr size_t OFF_XS   = OFF_WP   + SZ_WP;
static constexpr size_t OFF_PM   = OFF_XS   + SZ_XS;
static constexpr size_t OFF_UATT = OFF_PM   + SZ_PM;
static constexpr size_t OFF_UDEC = OFF_UATT + SZ_UATT;
static constexpr size_t OFF_HC   = OFF_UDEC + SZ_UDEC;
static constexpr size_t OFF_G    = OFF_HC   + SZ_HC;
static constexpr size_t OFF_ATTH = OFF_G    + SZ_G;
static constexpr size_t OFF_ATTC = OFF_ATTH + SZ_ST;
static constexpr size_t OFF_DECC = OFF_ATTC + SZ_ST;
static constexpr size_t OFF_BAR  = OFF_DECC + SZ_ST;

// ---------------------------------------------------------------------------
// Device-wide barrier for the persistent kernel (64 co-resident WGs)
// ---------------------------------------------------------------------------
__device__ __forceinline__ void grid_sync(unsigned* bar) {
    __syncthreads();
    if (threadIdx.x == 0) {
        volatile unsigned* vgen = bar + 1;
        unsigned gen = *vgen;
        __threadfence();
        if (atomicAdd(bar, 1u) == gridDim.x - 1) {
            *(volatile unsigned*)bar = 0;
            __threadfence();
            atomicAdd(bar + 1, 1u);
        } else {
            while (*vgen == gen) { __builtin_amdgcn_s_sleep(1); }
        }
        __threadfence();
    }
    __syncthreads();
}

// ---------------------------------------------------------------------------
// GEMM phase: gates(64x4096) = U(64xK) * W(4096xK)^T using WMMA bf16.
// 1024 tiles of 16x16 over 512 waves.
// ---------------------------------------------------------------------------
template <int K>
__device__ __forceinline__ void gemm_gates(const unsigned short* U,
                                           const unsigned short* W,
                                           float* G, int wglobal, int lane) {
    const int half = lane >> 4;
    const int lcol = lane & 15;
    for (int tile = wglobal; tile < 1024; tile += 512) {
        const int mt = tile & 3;
        const int nt = tile >> 2;
        const unsigned short* urow = U + (size_t)(mt * 16 + lcol) * K;
        const unsigned short* wrow = W + (size_t)(nt * 16 + lcol) * K;
        v8f acc = {0.f, 0.f, 0.f, 0.f, 0.f, 0.f, 0.f, 0.f};
        for (int kk = 0; kk < K; kk += 32) {
            v16bf a = load_a_frag(urow, kk, half);
            v16bf b = load_b_frag(wrow, kk, half);
            acc = __builtin_amdgcn_wmma_f32_16x16x32_bf16(
                false, a, false, b, (short)0, acc, false, false);
        }
        const int m0 = mt * 16 + half * 8;
        const int n  = nt * 16 + lcol;
#pragma unroll
        for (int r = 0; r < 8; ++r) G[(size_t)(m0 + r) * 4096 + n] = acc[r];
    }
}

// ---------------------------------------------------------------------------
// Precompute kernels
// ---------------------------------------------------------------------------
__global__ __launch_bounds__(256) void k_prep(
    const float* att_wih, const float* att_whh,
    const float* dec_wih, const float* dec_whh,
    const float* proj_w,  const float* gate_w,
    unsigned short* Wa, unsigned short* Wd, unsigned short* Wp) {
    const int NA = 4096 * 1792, ND = 4096 * 2560, NP = 96 * 1536;
    for (int i = blockIdx.x * blockDim.x + threadIdx.x; i < NA + ND + NP;
         i += gridDim.x * blockDim.x) {
        if (i < NA) {
            int n = i / 1792, k = i - n * 1792;
            float v = (k < 768) ? att_wih[(size_t)n * 768 + k]
                                : att_whh[(size_t)n * 1024 + (k - 768)];
            Wa[i] = f2bf(v);
        } else if (i < NA + ND) {
            int j = i - NA;
            int n = j / 2560, k = j - n * 2560;
            float v = (k < 1536) ? dec_wih[(size_t)n * 1536 + k]
                                 : dec_whh[(size_t)n * 1024 + (k - 1536)];
            Wd[j] = f2bf(v);
        } else {
            int j = i - NA - ND;
            int n = j / 1536, k = j - n * 1536;
            float v = (n < 80) ? proj_w[(size_t)n * 1536 + k]
                               : (n == 80 ? gate_w[k] : 0.f);
            Wp[j] = f2bf(v);
        }
    }
}

// Fused 2-layer prenet for all (step,batch): one WG per (s,b).
__global__ __launch_bounds__(256) void k_prenet(
    const float* dec_inputs, const float* w1, const float* w2,
    unsigned short* xs) {
    const int bid = blockIdx.x;              // s*64 + b
    const int s = bid >> 6, b = bid & 63;
    const int j = threadIdx.x;               // 0..255
    __shared__ float s_in[80];
    __shared__ float s_h1[256];
    if (j < 80) s_in[j] = (s == 0) ? 0.f
                                   : dec_inputs[(size_t)b * 80000 + (size_t)j * 1000 + (s - 1)];
    __syncthreads();
    float acc = 0.f;
    for (int m = 0; m < 80; ++m) acc += s_in[m] * w1[j * 80 + m];
    s_h1[j] = fmaxf(acc, 0.f);
    __syncthreads();
    float acc2 = 0.f;
    for (int m = 0; m < 256; ++m) acc2 += s_h1[m] * w2[j * 256 + m];
    xs[(size_t)s * 16384 + b * 256 + j] = f2bf(fmaxf(acc2, 0.f));
}

// processed_memory[b][t][a] = sum_e memory[b][t][e] * key_w[a][e]
__global__ __launch_bounds__(128) void k_pm(
    const float* memory, const float* key_w, float* pm) {
    const int bt = blockIdx.x;               // b*512 + t
    __shared__ float row[512];
    for (int e = threadIdx.x; e < 512; e += 128)
        row[e] = memory[(size_t)bt * 512 + e];
    __syncthreads();
    const int a = threadIdx.x;
    const float* kr = key_w + (size_t)a * 512;
    float acc = 0.f;
    for (int e = 0; e < 512; ++e) acc += row[e] * kr[e];
    pm[(size_t)bt * 128 + a] = acc;
}

// ---------------------------------------------------------------------------
// Persistent scan kernel: 64 WGs x 256 threads, 1000 sequential steps.
// ---------------------------------------------------------------------------
__global__ __launch_bounds__(256) void k_scan(
    const unsigned short* __restrict__ Wa, const unsigned short* __restrict__ Wd,
    const unsigned short* __restrict__ Wp, const unsigned short* __restrict__ xs,
    const float* __restrict__ pm, const float* __restrict__ mem,
    const int* __restrict__ mlen,
    const float* __restrict__ q_w, const float* __restrict__ conv_w,
    const float* __restrict__ dense_w, const float* __restrict__ v_w,
    const float* __restrict__ att_bih, const float* __restrict__ att_bhh,
    const float* __restrict__ dec_bih, const float* __restrict__ dec_bhh,
    const float* __restrict__ proj_b, const float* __restrict__ gate_b,
    unsigned short* u_att, unsigned short* u_dec, unsigned short* hc,
    float* G, float* att_h, float* att_c, float* dec_c,
    unsigned* bar, float* out_mel, float* out_gate, float* out_align) {

    __shared__ float s_aw[542];     // padded [15..526] live, conv reads [t..t+30]
    __shared__ float s_awc[542];
    __shared__ float s_pq[128];
    __shared__ float s_convw[32 * 62];
    __shared__ float s_dense[128 * 32];
    __shared__ float s_v[128];
    __shared__ float s_e[512];
    __shared__ float s_red[256];

    const int tid = threadIdx.x;
    const int bid = blockIdx.x;                 // == batch index for phase C
    const int gid = bid * 256 + tid;            // 0..16383
    const int lane = tid & 31;
    const int wglobal = bid * 8 + (tid >> 5);   // 0..511

    // ---- one-time init -----------------------------------------------------
    for (int i = tid; i < 542; i += 256) { s_aw[i] = 0.f; s_awc[i] = 0.f; }
    for (int i = tid; i < 32 * 62; i += 256) s_convw[i] = conv_w[i];
    for (int i = tid; i < 128 * 32; i += 256) s_dense[i] = dense_w[i];
    if (tid < 128) s_v[tid] = v_w[tid];
    for (int i = gid; i < 64 * 1792; i += 16384) u_att[i] = 0;  // xs[0] == 0 anyway
    for (int i = gid; i < 64 * 2560; i += 16384) u_dec[i] = 0;
    for (int i = gid; i < 64 * 1536; i += 16384) hc[i] = 0;
    for (int i = gid; i < 65536; i += 16384) {
        att_h[i] = 0.f; att_c[i] = 0.f; dec_c[i] = 0.f;
    }
    __syncthreads();
    grid_sync(bar);

    const int len = mlen[bid];

    for (int t = 0; t < 1000; ++t) {
        // ---- Phase A: attention-LSTM gates = u_att * Wa^T (WMMA) ----------
        gemm_gates<1792>(u_att, Wa, G, wglobal, lane);
        grid_sync(bar);

        // ---- Phase B: att-LSTM pointwise ----------------------------------
        for (int it = 0; it < 4; ++it) {
            const int idx = gid + it * 16384;           // 64*1024
            const int b = idx >> 10, j = idx & 1023;
            const float* g = G + (size_t)b * 4096;
            float vi = g[j]        + att_bih[j]        + att_bhh[j];
            float vf = g[1024 + j] + att_bih[1024 + j] + att_bhh[1024 + j];
            float vg = g[2048 + j] + att_bih[2048 + j] + att_bhh[2048 + j];
            float vo = g[3072 + j] + att_bih[3072 + j] + att_bhh[3072 + j];
            float c = sigmoidf_(vf) * att_c[idx] + sigmoidf_(vi) * tanhf(vg);
            float h = sigmoidf_(vo) * tanhf(c);
            att_c[idx] = c;
            att_h[idx] = h;
            unsigned short hb = f2bf(h);
            u_att[(size_t)b * 1792 + 768 + j] = hb;
            u_dec[(size_t)b * 2560 + j] = hb;
        }
        grid_sync(bar);

        // ---- Phase C: location attention (one WG per batch element) ------
        {
            const int b = bid;
            if (tid < 128) {                      // pq = att_h @ q_w^T
                const float* hrow = att_h + (size_t)b * 1024;
                const float* qr = q_w + (size_t)tid * 1024;
                float acc = 0.f;
                for (int j = 0; j < 1024; ++j) acc += hrow[j] * qr[j];
                s_pq[tid] = acc;
            }
            __syncthreads();

            for (int ht = 0; ht < 2; ++ht) {
                const int te = tid + ht * 256;
                float lf[32];
#pragma unroll
                for (int f = 0; f < 32; ++f) {
                    const float* w0 = &s_convw[f * 62];
                    float a = 0.f;
                    for (int k = 0; k < 31; ++k)
                        a += w0[k] * s_aw[te + k] + w0[31 + k] * s_awc[te + k];
                    lf[f] = a;
                }
                float e;
                if (te >= len) {
                    e = -__builtin_huge_valf();
                } else {
                    const float* pmr = pm + ((size_t)b * 512 + te) * 128;
                    float acc = 0.f;
                    for (int a = 0; a < 128; ++a) {
                        const float* dw = &s_dense[a * 32];
                        float pa = 0.f;
#pragma unroll
                        for (int f = 0; f < 32; ++f) pa += dw[f] * lf[f];
                        acc += s_v[a] * tanhf(s_pq[a] + pmr[a] + pa);
                    }
                    e = acc;
                }
                s_e[te] = e;
            }
            __syncthreads();

            // softmax over 512
            s_red[tid] = fmaxf(s_e[tid], s_e[tid + 256]);
            __syncthreads();
            for (int s = 128; s > 0; s >>= 1) {
                if (tid < s) s_red[tid] = fmaxf(s_red[tid], s_red[tid + s]);
                __syncthreads();
            }
            const float maxv = s_red[0];
            __syncthreads();
            float p0 = __expf(s_e[tid] - maxv);
            float p1 = __expf(s_e[tid + 256] - maxv);
            s_e[tid] = p0;
            s_e[tid + 256] = p1;
            s_red[tid] = p0 + p1;
            __syncthreads();
            for (int s = 128; s > 0; s >>= 1) {
                if (tid < s) s_red[tid] += s_red[tid + s];
                __syncthreads();
            }
            const float inv = 1.f / s_red[0];
            __syncthreads();

            for (int ht = 0; ht < 2; ++ht) {
                const int te = tid + ht * 256;
                float awv = s_e[te] * inv;
                s_aw[15 + te] = awv;
                s_awc[15 + te] += awv;
                out_align[((size_t)b * 1000 + t) * 512 + te] = awv;
            }
            __syncthreads();

            // ctx[b][e] = sum_t aw[t] * memory[b][t][e]
            for (int ht = 0; ht < 2; ++ht) {
                const int e = tid + ht * 256;
                const float* mb = mem + (size_t)b * 262144 + e;
                float acc = 0.f;
#pragma unroll 8
                for (int te = 0; te < 512; ++te)
                    acc += s_aw[15 + te] * mb[(size_t)te * 512];
                unsigned short cb = f2bf(acc);
                u_att[(size_t)b * 1792 + 256 + e] = cb;
                u_dec[(size_t)b * 2560 + 1024 + e] = cb;
                hc[(size_t)b * 1536 + 1024 + e] = cb;
            }
        }
        grid_sync(bar);

        // ---- Phase D: decoder-LSTM gates = u_dec * Wd^T (WMMA) -----------
        gemm_gates<2560>(u_dec, Wd, G, wglobal, lane);
        grid_sync(bar);

        // ---- Phase E: dec-LSTM pointwise ----------------------------------
        for (int it = 0; it < 4; ++it) {
            const int idx = gid + it * 16384;
            const int b = idx >> 10, j = idx & 1023;
            const float* g = G + (size_t)b * 4096;
            float vi = g[j]        + dec_bih[j]        + dec_bhh[j];
            float vf = g[1024 + j] + dec_bih[1024 + j] + dec_bhh[1024 + j];
            float vg = g[2048 + j] + dec_bih[2048 + j] + dec_bhh[2048 + j];
            float vo = g[3072 + j] + dec_bih[3072 + j] + dec_bhh[3072 + j];
            float c = sigmoidf_(vf) * dec_c[idx] + sigmoidf_(vi) * tanhf(vg);
            float h = sigmoidf_(vo) * tanhf(c);
            dec_c[idx] = c;
            unsigned short hb = f2bf(h);
            u_dec[(size_t)b * 2560 + 1536 + j] = hb;
            hc[(size_t)b * 1536 + j] = hb;
        }
        grid_sync(bar);

        // ---- Phase F: projection (WMMA) + stage next x_t ------------------
        if (wglobal < 24) {                       // 4 M-tiles x 6 N-tiles
            const int mt = wglobal & 3, nt = wglobal >> 2;
            const int half = lane >> 4, lcol = lane & 15;
            const unsigned short* urow = hc + (size_t)(mt * 16 + lcol) * 1536;
            const unsigned short* wrow = Wp + (size_t)(nt * 16 + lcol) * 1536;
            v8f acc = {0.f, 0.f, 0.f, 0.f, 0.f, 0.f, 0.f, 0.f};
            for (int kk = 0; kk < 1536; kk += 32) {
                v16bf a = load_a_frag(urow, kk, half);
                v16bf b = load_b_frag(wrow, kk, half);
                acc = __builtin_amdgcn_wmma_f32_16x16x32_bf16(
                    false, a, false, b, (short)0, acc, false, false);
            }
            const int n = nt * 16 + lcol;
            const int m0 = mt * 16 + half * 8;
            if (n < 80) {
                const float bi = proj_b[n];
#pragma unroll
                for (int r = 0; r < 8; ++r)
                    out_mel[(size_t)(m0 + r) * 80000 + (size_t)n * 1000 + t] = acc[r] + bi;
            } else if (n == 80) {
                const float bi = gate_b[0];
#pragma unroll
                for (int r = 0; r < 8; ++r)
                    out_gate[(size_t)(m0 + r) * 1000 + t] = acc[r] + bi;
            }
        }
        if (t + 1 < 1000) {                       // x_{t+1} into u_att[:,0:256]
            const int b = gid >> 8, j = gid & 255;
            u_att[(size_t)b * 1792 + j] = xs[(size_t)(t + 1) * 16384 + gid];
        }
        grid_sync(bar);
    }
}

// ---------------------------------------------------------------------------
// Host entry
// ---------------------------------------------------------------------------
extern "C" void kernel_launch(void* const* d_in, const int* in_sizes, int n_in,
                              void* d_out, int out_size, void* d_ws, size_t ws_size,
                              hipStream_t stream) {
    (void)in_sizes; (void)n_in; (void)out_size; (void)ws_size;

    const float* memory     = (const float*)d_in[0];
    const float* dec_inputs = (const float*)d_in[1];
    const int*   mlen       = (const int*)  d_in[2];
    const float* prenet_w1  = (const float*)d_in[3];
    const float* prenet_w2  = (const float*)d_in[4];
    const float* att_wih    = (const float*)d_in[5];
    const float* att_whh    = (const float*)d_in[6];
    const float* att_bih    = (const float*)d_in[7];
    const float* att_bhh    = (const float*)d_in[8];
    const float* q_w        = (const float*)d_in[9];
    const float* key_w      = (const float*)d_in[10];
    const float* conv_w     = (const float*)d_in[11];
    const float* dense_w    = (const float*)d_in[12];
    const float* v_w        = (const float*)d_in[13];
    const float* dec_wih    = (const float*)d_in[14];
    const float* dec_whh    = (const float*)d_in[15];
    const float* dec_bih    = (const float*)d_in[16];
    const float* dec_bhh    = (const float*)d_in[17];
    const float* proj_w     = (const float*)d_in[18];
    const float* proj_b     = (const float*)d_in[19];
    const float* gate_w     = (const float*)d_in[20];
    const float* gate_b     = (const float*)d_in[21];

    char* ws = (char*)d_ws;
    unsigned short* Wa   = (unsigned short*)(ws + OFF_WA);
    unsigned short* Wd   = (unsigned short*)(ws + OFF_WD);
    unsigned short* Wp   = (unsigned short*)(ws + OFF_WP);
    unsigned short* xs   = (unsigned short*)(ws + OFF_XS);
    float*          pm   = (float*)(ws + OFF_PM);
    unsigned short* uatt = (unsigned short*)(ws + OFF_UATT);
    unsigned short* udec = (unsigned short*)(ws + OFF_UDEC);
    unsigned short* hc   = (unsigned short*)(ws + OFF_HC);
    float*          G    = (float*)(ws + OFF_G);
    float*          atth = (float*)(ws + OFF_ATTH);
    float*          attc = (float*)(ws + OFF_ATTC);
    float*          decc = (float*)(ws + OFF_DECC);
    unsigned*       bar  = (unsigned*)(ws + OFF_BAR);

    float* out_mel   = (float*)d_out;
    float* out_gate  = out_mel + (size_t)64 * 80 * 1000;
    float* out_align = out_gate + (size_t)64 * 1000;

    hipMemsetAsync(ws + OFF_BAR, 0, 256, stream);

    k_prep<<<4096, 256, 0, stream>>>(att_wih, att_whh, dec_wih, dec_whh,
                                     proj_w, gate_w, Wa, Wd, Wp);
    k_prenet<<<64000, 256, 0, stream>>>(dec_inputs, prenet_w1, prenet_w2, xs);
    k_pm<<<32768, 128, 0, stream>>>(memory, key_w, pm);

    k_scan<<<64, 256, 0, stream>>>(Wa, Wd, Wp, xs, pm, memory, mlen,
                                   q_w, conv_w, dense_w, v_w,
                                   att_bih, att_bhh, dec_bih, dec_bhh,
                                   proj_b, gate_b,
                                   uatt, udec, hc, G, atth, attc, decc,
                                   bar, out_mel, out_gate, out_align);
}
